// PonderNet_11287174054614
// MI455X (gfx1250) — matrix operand order repaired and checked by reading, not compiled, not run on hardware
//
#include <hip/hip_runtime.h>

#define Bsz   32768
#define Ecnt  256
#define Hc    256
#define NSTEP 20
#define ROWS  64
#define NTHREADS 512
#define HPAD  (Hc + 8)     // bf16 h row pad (16B) -> conflict-free ds_load_b128
#define HPADF (Hc + 1)     // f32 h row pad

typedef __attribute__((ext_vector_type(16))) __bf16 v16bf;
typedef __attribute__((ext_vector_type(8)))  float  v8f;

struct alignas(32) Pack32 { uint4 lo, hi; };

static __device__ __forceinline__ unsigned short f2bf(float f) {
  unsigned u = __float_as_uint(f);
  u += 0x7FFFu + ((u >> 16) & 1u);          // round-to-nearest-even
  return (unsigned short)(u >> 16);
}

// Build a v16bf WMMA operand from two 16-byte chunks (global or LDS).
static __device__ __forceinline__ v16bf ld2(const unsigned short* p0,
                                            const unsigned short* p1) {
  Pack32 s;
  s.lo = *reinterpret_cast<const uint4*>(p0);
  s.hi = *reinterpret_cast<const uint4*>(p1);
  return __builtin_bit_cast(v16bf, s);
}

// ---------------- prep: x -> bf16 row major ----------------
__global__ void pn_cvt_x(const float* __restrict__ x,
                         unsigned short* __restrict__ xbf, int n) {
  int i = blockIdx.x * 256 + threadIdx.x;
  if (i < n) xbf[i] = f2bf(x[i]);
}

// ---------------- prep: pack W (768x256) into WMMA-B tile order ----------------
// element o = ((((kb*48 + nT)*2 + c)*32 + l)*8 + i)
// value    = bf16( W[ (nT*16 + (l&15)) * 256 + kb*32 + (l&16?8:0) + c*16 + i ] )
__global__ void pn_pack_w(const float* __restrict__ Wih,
                          const float* __restrict__ Whh,
                          unsigned short* __restrict__ pWih,
                          unsigned short* __restrict__ pWhh) {
  int o = blockIdx.x * 256 + threadIdx.x;       // 0..196607
  const float* W = blockIdx.y ? Whh : Wih;
  unsigned short* P = blockIdx.y ? pWhh : pWih;
  int i  = o & 7;
  int l  = (o >> 3) & 31;
  int c  = (o >> 8) & 1;
  int t9 = o >> 9;
  int nT = t9 % 48;
  int kb = t9 / 48;
  int j  = nT * 16 + (l & 15);
  int k  = kb * 32 + ((l & 16) ? 8 : 0) + c * 16 + i;
  P[o] = f2bf(W[j * 256 + k]);
}

// ---------------- main persistent kernel ----------------
// 64 batch rows per WG, 16 waves. Wave w owns hidden-column tile c=w (16 cols)
// for all 3 gates and all 4 M-tiles -> disjoint B tiles across waves (zero
// redundant W traffic), A operands shared via LDS.
__global__ __launch_bounds__(NTHREADS)
void pn_main(const float* __restrict__ bih,  const float* __restrict__ bhh,
             const float* __restrict__ Wout, const float* __restrict__ bout,
             const float* __restrict__ Wlam, const float* __restrict__ blam,
             const float* __restrict__ u,
             const unsigned short* __restrict__ xbf,
             const unsigned short* __restrict__ pWih,
             const unsigned short* __restrict__ pWhh,
             float* __restrict__ yOut, float* __restrict__ pOut,
             int* __restrict__ haltOut) {
  __shared__ float          lds_h[ROWS * HPADF];
  __shared__ unsigned short lds_hbf[ROWS * HPAD];
  __shared__ float          lds_wl[Hc], lds_wo[Hc];
  __shared__ float          lds_unh[ROWS];
  __shared__ int            lds_halt[ROWS];

  const int tid   = threadIdx.x;
  const int lane  = tid & 31;
  const int wid   = tid >> 5;             // 0..15 : owned hidden column tile
  const int lhalf = (lane >> 4) & 1;      // 0: lanes 0-15, 1: lanes 16-31
  const int lj    = lane & 15;
  const int jown  = wid * 16 + lj;        // owned hidden column (wave-uniform tile)
  const int rowBase = blockIdx.x * ROWS;

  if (tid < Hc) { lds_wl[tid] = Wlam[tid]; lds_wo[tid] = Wout[tid]; }
  if (tid < ROWS) { lds_unh[tid] = 1.0f; lds_halt[tid] = 0; }
  for (int i = tid; i < ROWS * HPADF; i += NTHREADS) lds_h[i]   = 0.0f;
  for (int i = tid; i < ROWS * HPAD;  i += NTHREADS) lds_hbf[i] = 0;
  __syncthreads();

  const v8f zero8 = {0.f,0.f,0.f,0.f,0.f,0.f,0.f,0.f};

  // ---- x-gates GEMM once: xg = x @ W_ih^T  (kept in registers) ----
  // index q = g*4 + mt  (gate g, M-tile mt)
  v8f xg[12];
  #pragma unroll
  for (int q = 0; q < 12; ++q) xg[q] = zero8;
  #pragma unroll 2
  for (int kb = 0; kb < 8; ++kb) {
    v16bf b[3];
    #pragma unroll
    for (int g = 0; g < 3; ++g) {
      const unsigned short* pb =
          pWih + (size_t)(kb * 48 + g * 16 + wid) * 512 + lane * 8;
      b[g] = ld2(pb, pb + 256);
    }
    v16bf a[4];
    #pragma unroll
    for (int mt = 0; mt < 4; ++mt) {
      const unsigned short* pa = xbf +
          (size_t)(rowBase + mt * 16 + lj) * Ecnt + kb * 32 + (lhalf ? 8 : 0);
      a[mt] = ld2(pa, pa + 16);
    }
    #pragma unroll
    for (int g = 0; g < 3; ++g)
      #pragma unroll
      for (int mt = 0; mt < 4; ++mt)
        xg[g * 4 + mt] = __builtin_amdgcn_wmma_f32_16x16x32_bf16(
            false, a[mt], false, b[g], (short)0, xg[g * 4 + mt], false, false);
  }
  // fold biases: b_ih (all gates) + b_hh (r,z); keep b_hh_n separate (scaled by r)
  const float bR = bih[jown]       + bhh[jown];
  const float bZ = bih[256 + jown] + bhh[256 + jown];
  const float bN = bih[512 + jown];
  const float bn = bhh[512 + jown];
  #pragma unroll
  for (int mt = 0; mt < 4; ++mt)
    #pragma unroll
    for (int v = 0; v < 8; ++v) {
      xg[mt][v]     += bR;
      xg[4 + mt][v] += bZ;
      xg[8 + mt][v] += bN;
    }

  v8f hv[4];                                   // owned h patch (fp32), per M-tile
  #pragma unroll
  for (int mt = 0; mt < 4; ++mt) hv[mt] = zero8;

  const float bO = bout[0], bL = blam[0];

  #pragma unroll 1
  for (int n = 1; n <= NSTEP; ++n) {
    // ---- hg = h @ W_hh^T : bf16 WMMA, fp32 accumulate ----
    v8f acc[12];
    #pragma unroll
    for (int q = 0; q < 12; ++q) acc[q] = zero8;
    #pragma unroll 2
    for (int kb = 0; kb < 8; ++kb) {
      v16bf b[3];
      #pragma unroll
      for (int g = 0; g < 3; ++g) {
        const unsigned short* pb =
            pWhh + (size_t)(kb * 48 + g * 16 + wid) * 512 + lane * 8;
        b[g] = ld2(pb, pb + 256);
      }
      v16bf a[4];
      #pragma unroll
      for (int mt = 0; mt < 4; ++mt) {
        const unsigned short* pa = lds_hbf +
            (mt * 16 + lj) * HPAD + kb * 32 + (lhalf ? 8 : 0);
        a[mt] = ld2(pa, pa + 16);
      }
      #pragma unroll
      for (int g = 0; g < 3; ++g)
        #pragma unroll
        for (int mt = 0; mt < 4; ++mt)
          acc[g * 4 + mt] = __builtin_amdgcn_wmma_f32_16x16x32_bf16(
              false, a[mt], false, b[g], (short)0, acc[g * 4 + mt], false, false);
    }

    // ---- heads (overlap with WMMA drain): lam, y from pre-update h ----
    {
      int m  = tid >> 3;                  // 0..63
      int k0 = (tid & 7) * 32;
      float sL = 0.f, sY = 0.f;
      #pragma unroll
      for (int i = 0; i < 32; ++i) {
        float hh = lds_h[m * HPADF + k0 + i];
        sL += hh * lds_wl[k0 + i];
        sY += hh * lds_wo[k0 + i];
      }
      sL += __shfl_xor(sL, 1, 8); sL += __shfl_xor(sL, 2, 8); sL += __shfl_xor(sL, 4, 8);
      sY += __shfl_xor(sY, 1, 8); sY += __shfl_xor(sY, 2, 8); sY += __shfl_xor(sY, 4, 8);
      if ((tid & 7) == 0) {
        int r = rowBase + m;
        float lam = 1.0f / (1.0f + __expf(-(sL + bL)));
        if (n == NSTEP) lam = 1.0f;
        float unh  = lds_unh[m];
        int   halt = lds_halt[m];
        float uv   = u[(size_t)(n - 1) * Bsz + r];
        yOut[(size_t)(n - 1) * Bsz + r] = sY + bO;
        pOut[(size_t)(n - 1) * Bsz + r] = unh * lam;
        if (halt == 0 && uv < lam) { halt = n; lds_halt[m] = halt; }
        lds_unh[m] = unh * (1.0f - lam);
        if (n == NSTEP) haltOut[r] = halt;
      }
    }

    __syncthreads();   // all reads of h / h_bf done before anyone rewrites

    // ---- GRU gates in registers; write back owned h patch ----
    #pragma unroll
    for (int mt = 0; mt < 4; ++mt) {
      #pragma unroll
      for (int v = 0; v < 8; ++v) {
        int m = mt * 16 + v + (lhalf ? 8 : 0);
        float r_ = 1.0f / (1.0f + __expf(-(xg[mt][v]     + acc[mt][v])));
        float z_ = 1.0f / (1.0f + __expf(-(xg[4 + mt][v] + acc[4 + mt][v])));
        float n_ = tanhf(xg[8 + mt][v] + r_ * (acc[8 + mt][v] + bn));
        float hn = (1.0f - z_) * n_ + z_ * hv[mt][v];
        hv[mt][v] = hn;
        lds_h[m * HPADF + jown]  = hn;
        lds_hbf[m * HPAD + jown] = f2bf(hn);
      }
    }
    __syncthreads();   // h updated before next step's readers
  }
}

extern "C" void kernel_launch(void* const* d_in, const int* in_sizes, int n_in,
                              void* d_out, int out_size, void* d_ws, size_t ws_size,
                              hipStream_t stream) {
  const float* x    = (const float*)d_in[0];
  const float* Wih  = (const float*)d_in[1];
  const float* Whh  = (const float*)d_in[2];
  const float* bih  = (const float*)d_in[3];
  const float* bhh  = (const float*)d_in[4];
  const float* Wout = (const float*)d_in[5];
  const float* bout = (const float*)d_in[6];
  const float* Wlam = (const float*)d_in[7];
  const float* blam = (const float*)d_in[8];
  const float* u    = (const float*)d_in[9];

  // workspace: packed W_ih (384KB) | packed W_hh (384KB) | x bf16 (16MB)
  char* ws = (char*)d_ws;
  unsigned short* pWih = (unsigned short*)ws;
  unsigned short* pWhh = (unsigned short*)(ws + 393216);
  unsigned short* xbf  = (unsigned short*)(ws + 2 * 393216);

  float* yOut    = (float*)d_out;
  float* pOut    = yOut + (size_t)NSTEP * Bsz;
  int*   haltOut = (int*)(yOut + (size_t)2 * NSTEP * Bsz);

  pn_cvt_x<<<(Bsz * Ecnt + 255) / 256, 256, 0, stream>>>(x, xbf, Bsz * Ecnt);
  dim3 gp(196608 / 256, 2);
  pn_pack_w<<<gp, 256, 0, stream>>>(Wih, Whh, pWih, pWhh);
  pn_main<<<Bsz / ROWS, NTHREADS, 0, stream>>>(bih, bhh, Wout, bout, Wlam, blam, u,
                                               xbf, pWih, pWhh, yOut, pOut, haltOut);
}
